// RRN_23888608101387
// MI455X (gfx1250) — compile-verified, matrix operand order checked by reading, not run on hardware
//
#include <hip/hip_runtime.h>
#include <hip/hip_bf16.h>
#include <math.h>

typedef __attribute__((ext_vector_type(16))) _Float16 v16h;
typedef __attribute__((ext_vector_type(8)))  float    v8f;

#define NROWS 200000
#define DDIM  128
#define KMEM  64
#define TTRI  50000
#define LLAY  8
#define KCLS  192   // D + K
#define KREL  256   // 2D
#define ITERS 2
#define SPAD  (TTRI + 512)   // sorted-array allocation (over-padded for tile prefetch)

// ---------------------------------------------------------------------------
// Fragment index helpers for V_WMMA_F32_16X16X32_F16 (wave32).
// A (16xK f16): lane l holds row m=l&15; halves j=0..7 -> k=j + (l>=16)*8,
//               j=8..15 -> k=16+(j-8)+(l>=16)*8 (per ISA 7.12.2).
// B (Kx16 f16): stored symmetrically (lane = column n, halves = k).
// C/D (16x16 f32, 8 VGPRs): c[i] = elem(m = i + 8*(l>=16), n = l&15).
// ---------------------------------------------------------------------------
__device__ __forceinline__ int frag_a_index(int m, int k) {
    int kstep = k >> 5;
    int kk    = k & 31;
    int hi    = (kk >> 3) & 1;
    int j     = (kk & 16) ? ((kk & 7) + 8) : (kk & 7);
    int lane  = m + hi * 16;
    return ((kstep * 32) + lane) * 16 + j;
}

__device__ __forceinline__ v8f wmma3(v16h ah, v16h al, v16h bh, v16h bl, v8f c) {
    // hi*lo cross terms recover ~f32 precision; lo*lo dropped (negligible)
    c = __builtin_amdgcn_wmma_f32_16x16x32_f16(false, al, false, bh, (short)0, c, false, false);
    c = __builtin_amdgcn_wmma_f32_16x16x32_f16(false, ah, false, bl, (short)0, c, false, false);
    c = __builtin_amdgcn_wmma_f32_16x16x32_f16(false, ah, false, bh, (short)0, c, false, false);
    return c;
}

// ---------------------------------------------------------------------------
// Triple bucketing by layer (done once; layer_idx is iteration-invariant)
// ---------------------------------------------------------------------------
__global__ void k_prep(int* counts, int* cursor, int* ssub, int* sobj) {
    int i = blockIdx.x * blockDim.x + threadIdx.x;
    if (i < LLAY) { counts[i] = 0; cursor[i] = 0; }
    if (i < SPAD) { ssub[i] = -1; sobj[i] = -1; }
}

__global__ void k_hist(const int* __restrict__ layer, int* counts) {
    int i = blockIdx.x * blockDim.x + threadIdx.x;
    if (i < TTRI) atomicAdd(&counts[layer[i]], 1);
}

__global__ void k_off(const int* __restrict__ counts, int* off) {
    if (threadIdx.x == 0 && blockIdx.x == 0) {
        int acc = 0;
        off[0] = 0;
        for (int l = 0; l < LLAY; ++l) {
            acc += ((counts[l] + 31) / 32) * 32;   // pad buckets to 32-row tiles
            off[l + 1] = acc;
        }
    }
}

__global__ void k_scatter(const int* __restrict__ layer, const int* __restrict__ subj,
                          const int* __restrict__ obj, const int* __restrict__ off,
                          int* cursor, int* ssub, int* sobj) {
    int i = blockIdx.x * blockDim.x + threadIdx.x;
    if (i < TTRI) {
        int l = layer[i];
        int p = off[l] + atomicAdd(&cursor[l], 1);
        ssub[p] = subj[i];
        sobj[p] = obj[i];
    }
}

// ---------------------------------------------------------------------------
// Weight splitting into fragment-ordered f16 hi/lo (done once)
// ---------------------------------------------------------------------------
__global__ void k_split_wc(const float* __restrict__ Wc, _Float16* wch, _Float16* wcl) {
    int idx = blockIdx.x * blockDim.x + threadIdx.x;
    if (idx >= 8 * 6 * 32 * 16) return;            // 8 ntiles * 6 ksteps * 512
    int j    = idx & 15;
    int lane = (idx >> 4) & 31;
    int r    = idx >> 9;
    int ks   = r % 6;
    int nt   = r / 6;
    int nl = lane & 15, hi = lane >> 4;
    int kk = ((j < 8) ? j : (j + 8)) + hi * 8;
    int k  = ks * 32 + kk;
    int n  = nt * 16 + nl;
    float w = Wc[k * DDIM + n];
    _Float16 h = (_Float16)w;
    wch[idx] = h;
    wcl[idx] = (_Float16)(w - (float)h);
}

__global__ void k_split_wr(const float* __restrict__ Wr, _Float16* wrh, _Float16* wrl) {
    int idx = blockIdx.x * blockDim.x + threadIdx.x;
    if (idx >= LLAY * 16 * 8 * 32 * 16) return;    // 8 layers * 16 ntiles * 8 ksteps * 512
    int j    = idx & 15;
    int lane = (idx >> 4) & 31;
    int r    = idx >> 9;                            // = l*128 + nt*8 + ks
    int ks   = r & 7;
    int nt   = (r >> 3) & 15;
    int l    = r >> 7;
    int nl = lane & 15, hi = lane >> 4;
    int kk = ((j < 8) ? j : (j + 8)) + hi * 8;
    int k  = ks * 32 + kk;
    int n  = nt * 16 + nl;
    float w = Wr[l * KREL * KREL + k * KREL + n];
    _Float16 h = (_Float16)w;
    wrh[idx] = h;
    wrl[idx] = (_Float16)(w - (float)h);
}

// ---------------------------------------------------------------------------
// Class update: e = l2norm(tanh([e; mem] @ Wc + bc)), in place (row-local).
// Block: 128 threads (4 waves), 32-row tile (2 M-subtiles), 8 N-tiles
// (2 per wave), 6 K-steps. Each B fragment pair feeds 6 WMMAs (2 M-subtiles).
// ---------------------------------------------------------------------------
__global__ __launch_bounds__(128) void k_class(
    float* __restrict__ e, const int* __restrict__ mem,
    const _Float16* __restrict__ wch, const _Float16* __restrict__ wcl,
    const float* __restrict__ bc)
{
    __shared__ alignas(32) _Float16 sAh[2 * 6 * 32 * 16];
    __shared__ alignas(32) _Float16 sAl[2 * 6 * 32 * 16];
    __shared__ float sC[32 * DDIM];
    __shared__ float sSq[32];

    const int tid  = threadIdx.x;
    const int lane = tid & 31;
    const int wave = tid >> 5;
    const int row0 = blockIdx.x * 32;

    // stage A tile (hi/lo split) into fragment-ordered LDS
    for (int idx = tid; idx < 32 * KCLS; idx += 128) {
        int m = idx / KCLS;
        int c = idx - m * KCLS;
        int r = row0 + m;
        float v = (c < DDIM) ? e[r * DDIM + c] : (float)mem[r * KMEM + (c - DDIM)];
        _Float16 h = (_Float16)v;
        int fi = (m >> 4) * (6 * 512) + frag_a_index(m & 15, c);
        sAh[fi] = h;
        sAl[fi] = (_Float16)(v - (float)h);
    }
    __syncthreads();

    v8f acc[2][2] = {{{}, {}}, {{}, {}}};    // [mt][nt]
    const int nt0 = wave * 2;
    for (int ks = 0; ks < 6; ++ks) {
        v16h ah0 = *(const v16h*)&sAh[(ks * 32 + lane) * 16];
        v16h al0 = *(const v16h*)&sAl[(ks * 32 + lane) * 16];
        v16h ah1 = *(const v16h*)&sAh[(6 * 512) + (ks * 32 + lane) * 16];
        v16h al1 = *(const v16h*)&sAl[(6 * 512) + (ks * 32 + lane) * 16];
        for (int q = 0; q < 2; ++q) {
            int base = (((nt0 + q) * 6 + ks) * 32 + lane) * 16;
            v16h bh = *(const v16h*)&wch[base];
            v16h bl = *(const v16h*)&wcl[base];
            acc[0][q] = wmma3(ah0, al0, bh, bl, acc[0][q]);
            acc[1][q] = wmma3(ah1, al1, bh, bl, acc[1][q]);
        }
    }

    // epilogue: bias + tanh -> LDS
    {
        int hi = lane >> 4, nl = lane & 15;
        for (int q = 0; q < 2; ++q) {
            int n = (nt0 + q) * 16 + nl;
            float b = bc[n];
            for (int mt = 0; mt < 2; ++mt)
                for (int i = 0; i < 8; ++i) {
                    int m = mt * 16 + i + 8 * hi;
                    sC[m * DDIM + n] = tanhf(acc[mt][q][i] + b);
                }
        }
    }
    __syncthreads();
    if (tid < 32) sSq[tid] = 0.f;
    __syncthreads();
    {
        int m = tid >> 2, seg = tid & 3;          // 4 threads/row, 32 cols each
        float s = 0.f;
        for (int c = seg * 32; c < seg * 32 + 32; ++c) { float v = sC[m * DDIM + c]; s += v * v; }
        atomicAdd(&sSq[m], s);
    }
    __syncthreads();
    for (int idx = tid; idx < 32 * DDIM; idx += 128) {
        int m = idx >> 7, c = idx & 127;
        float nrm = fmaxf(sqrtf(sSq[m]), 1e-12f);
        e[(row0 + m) * DDIM + c] = sC[idx] / nrm;
    }
}

// ---------------------------------------------------------------------------
// Relation update: per layer-bucketed 32-triple tile,
// out = tanh([e[s];e[o]] @ Wr[l] + br[l]); e[s]=l2norm(out[:128]); e[o]=l2norm(out[128:])
// Block: 128 threads (4 waves), 32-row tile (2 M-subtiles), 16 N-tiles
// (4 per wave), 8 K-steps. In place (all 2T scattered rows are distinct).
// ---------------------------------------------------------------------------
__global__ __launch_bounds__(128) void k_rel(
    float* __restrict__ e,
    const int* __restrict__ ssub, const int* __restrict__ sobj,
    const int* __restrict__ off,
    const _Float16* __restrict__ wrh, const _Float16* __restrict__ wrl,
    const float* __restrict__ br)
{
    __shared__ alignas(32) _Float16 sAh[2 * 8 * 32 * 16];
    __shared__ alignas(32) _Float16 sAl[2 * 8 * 32 * 16];
    __shared__ float sC[32 * KREL];
    __shared__ float sSq[2][32];
    __shared__ int   sSub[32], sObj[32], sLayer;

    const int tid  = threadIdx.x;
    const int lane = tid & 31;
    const int wave = tid >> 5;
    const int row0 = blockIdx.x * 32;

    if (tid == 0) {
        int l = -1;
        if (row0 < off[LLAY]) {
            l = 0;
            while (l < LLAY - 1 && row0 >= off[l + 1]) ++l;
        }
        sLayer = l;
    }
    if (tid < 32) {
        sSub[tid] = ssub[row0 + tid];   // arrays allocated SPAD, -1 padded
        sObj[tid] = sobj[row0 + tid];
    }
    __syncthreads();
    const int layer = sLayer;
    if (layer < 0) return;

    // gather pair rows, split, stage fragment-ordered
    for (int idx = tid; idx < 32 * KREL; idx += 128) {
        int m = idx >> 8, c = idx & 255;
        int r = (c < DDIM) ? sSub[m] : sObj[m];
        float v = (r >= 0) ? e[r * DDIM + (c & 127)] : 0.f;
        _Float16 h = (_Float16)v;
        int fi = (m >> 4) * (8 * 512) + frag_a_index(m & 15, c);
        sAh[fi] = h;
        sAl[fi] = (_Float16)(v - (float)h);
    }
    __syncthreads();

    v8f acc[2][4] = {{{}, {}, {}, {}}, {{}, {}, {}, {}}};   // [mt][nt]
    for (int ks = 0; ks < 8; ++ks) {
        v16h ah0 = *(const v16h*)&sAh[(ks * 32 + lane) * 16];
        v16h al0 = *(const v16h*)&sAl[(ks * 32 + lane) * 16];
        v16h ah1 = *(const v16h*)&sAh[(8 * 512) + (ks * 32 + lane) * 16];
        v16h al1 = *(const v16h*)&sAl[(8 * 512) + (ks * 32 + lane) * 16];
        for (int q = 0; q < 4; ++q) {
            int nt = wave * 4 + q;
            int base = (((layer * 16 + nt) * 8 + ks) * 32 + lane) * 16;
            v16h bh = *(const v16h*)&wrh[base];
            v16h bl = *(const v16h*)&wrl[base];
            acc[0][q] = wmma3(ah0, al0, bh, bl, acc[0][q]);
            acc[1][q] = wmma3(ah1, al1, bh, bl, acc[1][q]);
        }
    }

    {
        int hi = lane >> 4, nl = lane & 15;
        for (int q = 0; q < 4; ++q) {
            int n = (wave * 4 + q) * 16 + nl;
            float b = br[layer * KREL + n];
            for (int mt = 0; mt < 2; ++mt)
                for (int i = 0; i < 8; ++i) {
                    int m = mt * 16 + i + 8 * hi;
                    sC[m * KREL + n] = tanhf(acc[mt][q][i] + b);
                }
        }
    }
    __syncthreads();
    if (tid < 64) ((float*)sSq)[tid] = 0.f;
    __syncthreads();
    {
        int m = tid >> 2, seg = tid & 3;          // 4 threads/row, 64 cols each
        float s = 0.f;
        for (int c = seg * 64; c < seg * 64 + 64; ++c) { float v = sC[m * KREL + c]; s += v * v; }
        atomicAdd(&sSq[seg >> 1][m], s);
    }
    __syncthreads();
    for (int idx = tid; idx < 32 * KREL; idx += 128) {
        int m = idx >> 8, c = idx & 255;
        if (sSub[m] < 0) continue;                // padding row
        int half = c >> 7;
        int r = half ? sObj[m] : sSub[m];
        float nrm = fmaxf(sqrtf(sSq[half][m]), 1e-12f);
        e[r * DDIM + (c & 127)] = sC[idx] / nrm;
    }
}

// ---------------------------------------------------------------------------
static inline size_t align_up(size_t x, size_t a) { return (x + a - 1) & ~(a - 1); }

extern "C" void kernel_launch(void* const* d_in, const int* in_sizes, int n_in,
                              void* d_out, int out_size, void* d_ws, size_t ws_size,
                              hipStream_t stream) {
    const float* emb   = (const float*)d_in[0];
    const int*   mem   = (const int*)d_in[1];
    const int*   subj  = (const int*)d_in[2];
    const int*   obj   = (const int*)d_in[3];
    const int*   layer = (const int*)d_in[4];
    const float* Wc    = (const float*)d_in[5];
    const float* bc    = (const float*)d_in[6];
    const float* Wr    = (const float*)d_in[7];
    const float* br    = (const float*)d_in[8];
    float* e = (float*)d_out;

    // workspace layout
    char* w = (char*)d_ws;
    size_t p = 0;
    int* counts = (int*)(w + p); p += 8 * sizeof(int);
    int* cursor = (int*)(w + p); p += 8 * sizeof(int);
    int* off    = (int*)(w + p); p += 9 * sizeof(int);
    p = align_up(p, 256);
    int* ssub = (int*)(w + p); p += (size_t)SPAD * sizeof(int);
    int* sobj = (int*)(w + p); p += (size_t)SPAD * sizeof(int);
    p = align_up(p, 256);
    _Float16* wch = (_Float16*)(w + p); p += (size_t)KCLS * DDIM * sizeof(_Float16);
    p = align_up(p, 256);
    _Float16* wcl = (_Float16*)(w + p); p += (size_t)KCLS * DDIM * sizeof(_Float16);
    p = align_up(p, 256);
    _Float16* wrh = (_Float16*)(w + p); p += (size_t)LLAY * KREL * KREL * sizeof(_Float16);
    p = align_up(p, 256);
    _Float16* wrl = (_Float16*)(w + p); p += (size_t)LLAY * KREL * KREL * sizeof(_Float16);
    (void)ws_size; (void)in_sizes; (void)n_in; (void)out_size;

    // e buffer <- initial embeddings (in-place pipeline thereafter)
    hipMemcpyAsync(e, emb, (size_t)NROWS * DDIM * sizeof(float),
                   hipMemcpyDeviceToDevice, stream);

    // bucket triples by layer (iteration-invariant)
    k_prep<<<(SPAD + 255) / 256, 256, 0, stream>>>(counts, cursor, ssub, sobj);
    k_hist<<<(TTRI + 255) / 256, 256, 0, stream>>>(layer, counts);
    k_off<<<1, 1, 0, stream>>>(counts, off);
    k_scatter<<<(TTRI + 255) / 256, 256, 0, stream>>>(layer, subj, obj, off, cursor, ssub, sobj);

    // split weights into fragment-ordered f16 hi/lo
    k_split_wc<<<(KCLS * DDIM + 255) / 256, 256, 0, stream>>>(Wc, wch, wcl);
    k_split_wr<<<(LLAY * KREL * KREL + 255) / 256, 256, 0, stream>>>(Wr, wrh, wrl);

    const int relBlocks = (TTRI + 31) / 32 + LLAY;
    for (int it = 0; it < ITERS; ++it) {
        k_class<<<NROWS / 32, 128, 0, stream>>>(e, mem, wch, wcl, bc);
        k_rel<<<relBlocks, 128, 0, stream>>>(e, ssub, sobj, off, wrh, wrl, br);
    }
}
